// GPTModel_52037823758378
// MI455X (gfx1250) — compile-verified, matrix operand order checked by reading, not run on hardware
//
#include <hip/hip_runtime.h>
#include <hip/hip_bf16.h>

// ---- model dims (match reference) ----
constexpr int kB  = 2;
constexpr int kT  = 1024;
constexpr int kV  = 32000;
constexpr int kD  = 1024;
constexpr int kH  = 16;
constexpr int kL  = 8;
constexpr int kDF = 4096;
constexpr int kHD = 64;

typedef __attribute__((ext_vector_type(16))) __bf16 v16bf;
typedef __attribute__((ext_vector_type(8)))  __bf16 bf16x8;
typedef __attribute__((ext_vector_type(8)))  float  v8f;

// ---------------------------------------------------------------------------
// Embedding + sinusoidal positional encoding -> h (fp32 residual stream)
// ---------------------------------------------------------------------------
__global__ void embed_pe_kernel(const int* __restrict__ x,
                                const float* __restrict__ W_tok,
                                float* __restrict__ h) {
    int idx = blockIdx.x * blockDim.x + threadIdx.x;   // over B*T*D
    int d  = idx % kD;
    int bt = idx / kD;
    int t  = bt % kT;
    int tok = x[bt];
    float v = (tok == 0) ? 0.0f : W_tok[(long)tok * kD + d];
    float e   = (float)(d & ~1) * (-9.210340371976184f / (float)kD);
    float ang = (float)t * expf(e);
    v += (d & 1) ? cosf(ang) : sinf(ang);
    h[idx] = v;
}

// ---------------------------------------------------------------------------
// Transpose + fp32->bf16 convert:  src[R][C] fp32  ->  dst[C][R] bf16
// ---------------------------------------------------------------------------
__global__ __launch_bounds__(256) void transpose_convert_kernel(
    const float* __restrict__ src, __bf16* __restrict__ dst, int R, int C) {
    __shared__ float tile[32][33];
    const int tx = threadIdx.x & 31, ty = threadIdx.x >> 5;   // 32 x 8
    const int r0 = blockIdx.y * 32, c0 = blockIdx.x * 32;
#pragma unroll
    for (int j = 0; j < 32; j += 8)
        tile[ty + j][tx] = src[(long)(r0 + ty + j) * C + c0 + tx];
    __syncthreads();
#pragma unroll
    for (int j = 0; j < 32; j += 8)
        dst[(long)(c0 + ty + j) * R + r0 + tx] = (__bf16)tile[tx][ty + j];
}

// ---------------------------------------------------------------------------
// Per-(b,h) transpose of V (bf16): qkv[b,t,2D+h*HD+hd] -> Vt[b,h,hd,t]
// ---------------------------------------------------------------------------
__global__ __launch_bounds__(256) void transpose_v_kernel(
    const __bf16* __restrict__ qkv, __bf16* __restrict__ Vt) {
    const int bh = blockIdx.z;
    const int b = bh / kH, hh = bh % kH;
    const __bf16* src = qkv + (long)b * kT * 3 * kD + 2 * kD + hh * kHD; // [t][hd], stride 3D
    __bf16* dst = Vt + (long)bh * kHD * kT;                               // [hd][t], stride T
    __shared__ __bf16 tile[32][33];
    const int tx = threadIdx.x & 31, ty = threadIdx.x >> 5;
    const int t0 = blockIdx.x * 32, hd0 = blockIdx.y * 32;
#pragma unroll
    for (int j = 0; j < 32; j += 8)
        tile[ty + j][tx] = src[(long)(t0 + ty + j) * 3 * kD + hd0 + tx];
    __syncthreads();
#pragma unroll
    for (int j = 0; j < 32; j += 8)
        dst[(long)(hd0 + ty + j) * kT + t0 + tx] = tile[tx][ty + j];
}

// ---------------------------------------------------------------------------
// fp32 -> bf16 elementwise (residual stream -> GEMM A operand)
// ---------------------------------------------------------------------------
__global__ void cvt_f32_bf16_kernel(const float* __restrict__ s, __bf16* __restrict__ d) {
    int i = blockIdx.x * blockDim.x + threadIdx.x;
    d[i] = (__bf16)s[i];
}

// ---------------------------------------------------------------------------
// LayerNorm: fp32 in, bf16 out (consumed only as GEMM A operand)
// ---------------------------------------------------------------------------
__global__ __launch_bounds__(256) void layernorm_kernel(
    const float* __restrict__ x, const float* __restrict__ w,
    const float* __restrict__ b, __bf16* __restrict__ y) {
    const int row = blockIdx.x;
    const float* xr = x + (long)row * kD;
    float s = 0.f, s2 = 0.f;
    for (int i = threadIdx.x; i < kD; i += 256) {
        float v = xr[i];
        s += v; s2 += v * v;
    }
    __shared__ float r1[256], r2[256];
    r1[threadIdx.x] = s; r2[threadIdx.x] = s2;
    __syncthreads();
    for (int off = 128; off > 0; off >>= 1) {
        if (threadIdx.x < off) {
            r1[threadIdx.x] += r1[threadIdx.x + off];
            r2[threadIdx.x] += r2[threadIdx.x + off];
        }
        __syncthreads();
    }
    float mu  = r1[0] * (1.0f / kD);
    float var = r2[0] * (1.0f / kD) - mu * mu;
    float inv = rsqrtf(var + 1e-5f);
    for (int i = threadIdx.x; i < kD; i += 256)
        y[(long)row * kD + i] = (__bf16)((xr[i] - mu) * inv * w[i] + b[i]);
}

// ---------------------------------------------------------------------------
// Causal softmax: S fp32 [B*H*T rows, T] -> P bf16 (masked tail zeroed)
// ---------------------------------------------------------------------------
__global__ __launch_bounds__(256) void softmax_causal_kernel(
    const float* __restrict__ S, __bf16* __restrict__ P) {
    const int row = blockIdx.x;
    const int q   = row % kT;
    const int n   = q + 1;
    const float* sr = S + (long)row * kT;
    __bf16* pr = P + (long)row * kT;
    __shared__ float red[256];

    float mx = -__builtin_inff();
    for (int j = threadIdx.x; j < n; j += 256) mx = fmaxf(mx, sr[j]);
    red[threadIdx.x] = mx;
    __syncthreads();
    for (int off = 128; off > 0; off >>= 1) {
        if (threadIdx.x < off) red[threadIdx.x] = fmaxf(red[threadIdx.x], red[threadIdx.x + off]);
        __syncthreads();
    }
    mx = red[0];
    __syncthreads();

    float sum = 0.f;
    for (int j = threadIdx.x; j < n; j += 256) sum += expf(sr[j] - mx);
    red[threadIdx.x] = sum;
    __syncthreads();
    for (int off = 128; off > 0; off >>= 1) {
        if (threadIdx.x < off) red[threadIdx.x] += red[threadIdx.x + off];
        __syncthreads();
    }
    float rinv = 1.0f / red[0];
    for (int j = threadIdx.x; j < kT; j += 256)
        pr[j] = (j < n) ? (__bf16)(expf(sr[j] - mx) * rinv) : (__bf16)0.0f;
}

__device__ __forceinline__ float gelu_exact(float x) {
    return 0.5f * x * (1.0f + erff(x * 0.7071067811865476f));
}

// ---------------------------------------------------------------------------
// Batched bf16 WMMA GEMM, epilogue specialized at compile time.
//   A: bf16 [M][K] k-contiguous (lda);  B: bf16 [N][K] k-contiguous (ldb)
//   C: fp32 or bf16 [M][N] (ldc)
//   C[m,n] = act( alpha * sum_k A[m,k]*B[n,k] [+ bias[n]] [+ R[m,n]] )
// Batch z = b*Hinner + h.  M%BM==0, N%BN==0, K%32==0, lda/ldb%8==0.
// 256 threads = 8 waves (4x2). Wave tile = (BM/4) x (BN/2).
// ---------------------------------------------------------------------------
template <int BM, int BN, bool HAS_BIAS, bool HAS_RES, int ACT, bool C_BF16>
__global__ __launch_bounds__(256) void gemm_wmma_bf16(
    const __bf16* __restrict__ A, long lda, long sAb, long sAh,
    const __bf16* __restrict__ Bt, long ldb, long sBb, long sBh,
    void* __restrict__ Cp, long ldc, long sCb, long sCh,
    const float* __restrict__ bias,
    const float* __restrict__ R, long ldr, long sRb, long sRh,
    float alpha, int K, int Hinner, int causalSkip, int clipK) {
    if (causalSkip && (int)blockIdx.x > (int)blockIdx.y) return;

    constexpr int WM = BM / 4;       // wave tile rows
    constexpr int WN = BN / 2;       // wave tile cols
    constexpr int FM = WM / 16;      // A fragments per wave
    constexpr int FN = WN / 16;      // B fragments per wave
    constexpr int CHA = BM / 64;     // 16B staging chunks per thread (A)
    constexpr int CHB = BN / 64;     // 16B staging chunks per thread (B)
    constexpr int PITCH = 48;        // bf16 LDS pitch: 96B rows, 16B-aligned slices

    const int bz = blockIdx.z;
    const int bb = bz / Hinner;
    const int hh = bz % Hinner;
    A  += (long)bb * sAb + (long)hh * sAh;
    Bt += (long)bb * sBb + (long)hh * sBh;
    if (HAS_RES) R += (long)bb * sRb + (long)hh * sRh;

    const int m0 = blockIdx.y * BM;
    const int n0 = blockIdx.x * BN;
    int Keff = K;
    if (clipK) { int kl = m0 + BM; Keff = (kl < K) ? kl : K; }   // P@V: P zero past diag

    __shared__ __align__(16) __bf16 As[BM * PITCH];
    __shared__ __align__(16) __bf16 Bs[BN * PITCH];

    const int tid  = threadIdx.x;
    const int lane = tid & 31;
    const int wv   = tid >> 5;
    const int mw   = wv >> 1;        // 0..3
    const int nw   = wv & 1;         // 0..1
    const int ln   = lane & 15;
    const int half = lane >> 4;

    // per-thread staging bases: row = tid>>2 (+64 per chunk), col slice = (tid&3)*8
    const int srow = tid >> 2;
    const int scol = (tid & 3) * 8;
    const __bf16* pA = A + (long)(m0 + srow) * lda + scol;
    const __bf16* pB = Bt + (long)(n0 + srow) * ldb + scol;
    const int ldsA = srow * PITCH + scol;
    const int ldsB = srow * PITCH + scol;

    v8f acc[FM][FN];
#pragma unroll
    for (int i = 0; i < FM; ++i)
#pragma unroll
        for (int j = 0; j < FN; ++j) acc[i][j] = (v8f){};

    for (int k0 = 0; k0 < Keff; k0 += 32) {
        // load staging chunks into registers (overlaps with previous compute)
        uint4 ra[CHA], rb[CHB];
#pragma unroll
        for (int i = 0; i < CHA; ++i)
            ra[i] = *(const uint4*)(pA + (long)i * 64 * lda + k0);
#pragma unroll
        for (int i = 0; i < CHB; ++i)
            rb[i] = *(const uint4*)(pB + (long)i * 64 * ldb + k0);
        if (k0 + 32 < Keff) {   // speculative prefetch of next K tile
            __builtin_prefetch(pA + k0 + 32, 0, 1);
            __builtin_prefetch(pB + k0 + 32, 0, 1);
        }
        __syncthreads();
#pragma unroll
        for (int i = 0; i < CHA; ++i) *(uint4*)&As[ldsA + i * 64 * PITCH] = ra[i];
#pragma unroll
        for (int i = 0; i < CHB; ++i) *(uint4*)&Bs[ldsB + i * 64 * PITCH] = rb[i];
        __syncthreads();

        // A fragments (ISA 16-bit layout: half 0 -> K 0..7,16..23; half 1 -> K 8..15,24..31)
        v16bf af[FM];
#pragma unroll
        for (int fm = 0; fm < FM; ++fm) {
            const __bf16* pa = &As[(mw * WM + fm * 16 + ln) * PITCH];
            bf16x8 lo = *(const bf16x8*)(pa + half * 8);
            bf16x8 hi = *(const bf16x8*)(pa + 16 + half * 8);
#pragma unroll
            for (int e = 0; e < 8; ++e) { af[fm][e] = lo[e]; af[fm][8 + e] = hi[e]; }
        }
#pragma unroll
        for (int fn = 0; fn < FN; ++fn) {
            const __bf16* pb = &Bs[(nw * WN + fn * 16 + ln) * PITCH];
            bf16x8 lo = *(const bf16x8*)(pb + half * 8);
            bf16x8 hi = *(const bf16x8*)(pb + 16 + half * 8);
            v16bf bf;
#pragma unroll
            for (int e = 0; e < 8; ++e) { bf[e] = lo[e]; bf[8 + e] = hi[e]; }
#pragma unroll
            for (int fm = 0; fm < FM; ++fm)
                acc[fm][fn] = __builtin_amdgcn_wmma_f32_16x16x32_bf16(
                    false, af[fm], false, bf, (short)0, acc[fm][fn], false, false);
        }
    }

    // epilogue (straight-line; all feature flags compile-time)
    float* Cf = (float*)Cp;
    __bf16* Ch = (__bf16*)Cp;
    const long cBase = (long)bb * sCb + (long)hh * sCh;
#pragma unroll
    for (int fm = 0; fm < FM; ++fm)
#pragma unroll
        for (int r = 0; r < 8; ++r) {
            const int m = m0 + mw * WM + fm * 16 + half * 8 + r;
            const long crow = cBase + (long)m * ldc;
            const float* Rrow = HAS_RES ? (R + (long)m * ldr) : nullptr;
#pragma unroll
            for (int fn = 0; fn < FN; ++fn) {
                const int n = n0 + nw * WN + fn * 16 + ln;
                float v = acc[fm][fn][r] * alpha;
                if (HAS_BIAS) v += bias[n];
                if (HAS_RES)  v += Rrow[n];
                if (ACT == 1) v = gelu_exact(v);
                if (C_BF16) Ch[crow + n] = (__bf16)v;
                else        Cf[crow + n] = v;
            }
        }
}

// ---------------------------------------------------------------------------
extern "C" void kernel_launch(void* const* d_in, const int* in_sizes, int n_in,
                              void* d_out, int out_size, void* d_ws, size_t ws_size,
                              hipStream_t stream) {
    (void)in_sizes; (void)n_in; (void)out_size; (void)ws_size;
    const int*   x      = (const int*)  d_in[0];
    const float* W_tok  = (const float*)d_in[1];
    const float* ln1_w  = (const float*)d_in[2];
    const float* ln1_b  = (const float*)d_in[3];
    const float* attn_w = (const float*)d_in[4];
    const float* attn_b = (const float*)d_in[5];
    const float* proj_w = (const float*)d_in[6];
    const float* proj_b = (const float*)d_in[7];
    const float* ln2_w  = (const float*)d_in[8];
    const float* ln2_b  = (const float*)d_in[9];
    const float* ff1_w  = (const float*)d_in[10];
    const float* ff1_b  = (const float*)d_in[11];
    const float* ff2_w  = (const float*)d_in[12];
    const float* ff2_b  = (const float*)d_in[13];
    const float* W_out  = (const float*)d_in[14];
    float* out = (float*)d_out;

    const long BT = (long)kB * kT;

    // workspace carve (256B aligned regions)
    size_t off = 0;
    char* base = (char*)d_ws;
    auto carve = [&](size_t bytes) -> void* {
        void* p = base + off;
        off += (bytes + 255) & ~(size_t)255;
        return p;
    };
    float*  hbuf = (float*) carve(BT * kD * 4);                 // fp32 residual stream
    __bf16* hb16 = (__bf16*)carve(BT * kD * 2);                 // h in bf16 (logits A)
    __bf16* ubf  = (__bf16*)carve(BT * kD * 2);                 // LN output
    __bf16* qkvb = (__bf16*)carve(BT * 3 * kD * 2);             // QKV (bf16)
    float*  Sbuf = (float*) carve((long)kB * kH * kT * kT * 4); // scores fp32
    __bf16* Pb   = (__bf16*)carve((long)kB * kH * kT * kT * 2); // probs bf16
    __bf16* Vt   = (__bf16*)carve((long)kB * kH * kHD * kT * 2);// V transposed
    __bf16* obf  = (__bf16*)carve(BT * kD * 2);                 // attn out
    __bf16* ffb  = (__bf16*)carve(BT * kDF * 2);                // FFN hidden
    __bf16* attn_wt = (__bf16*)carve((long)kL * 3 * kD * kD * 2);
    __bf16* proj_wt = (__bf16*)carve((long)kL * kD * kD * 2);
    __bf16* ff1_wt  = (__bf16*)carve((long)kL * kDF * kD * 2);
    __bf16* ff2_wt  = (__bf16*)carve((long)kL * kD * kDF * 2);
    __bf16* wout_t  = (__bf16*)carve((long)kV * kD * 2);

    // ---- one-time weight transpose+convert to bf16 [N][K] ----
    for (int l = 0; l < kL; ++l) {
        transpose_convert_kernel<<<dim3(3 * kD / 32, kD / 32), 256, 0, stream>>>(
            attn_w + (long)l * kD * 3 * kD, attn_wt + (long)l * 3 * kD * kD, kD, 3 * kD);
        transpose_convert_kernel<<<dim3(kD / 32, kD / 32), 256, 0, stream>>>(
            proj_w + (long)l * kD * kD, proj_wt + (long)l * kD * kD, kD, kD);
        transpose_convert_kernel<<<dim3(kDF / 32, kD / 32), 256, 0, stream>>>(
            ff1_w + (long)l * kD * kDF, ff1_wt + (long)l * kDF * kD, kD, kDF);
        transpose_convert_kernel<<<dim3(kD / 32, kDF / 32), 256, 0, stream>>>(
            ff2_w + (long)l * kDF * kD, ff2_wt + (long)l * kD * kDF, kDF, kD);
    }
    transpose_convert_kernel<<<dim3(kV / 32, kD / 32), 256, 0, stream>>>(
        W_out, wout_t, kD, kV);

    // ---- embedding ----
    embed_pe_kernel<<<(kB * kT * kD) / 256, 256, 0, stream>>>(x, W_tok, hbuf);

    for (int l = 0; l < kL; ++l) {
        // u = LN1(h)  (bf16)
        layernorm_kernel<<<BT, 256, 0, stream>>>(
            hbuf, ln1_w + (long)l * kD, ln1_b + (long)l * kD, ubf);

        // qkv = u @ attn_w + attn_b   -> bf16 [BT,3D]
        gemm_wmma_bf16<128, 128, true, false, 0, true>
            <<<dim3(3 * kD / 128, BT / 128, 1), 256, 0, stream>>>(
            ubf, kD, 0, 0,
            attn_wt + (long)l * 3 * kD * kD, kD, 0, 0,
            qkvb, 3 * kD, 0, 0,
            attn_b + (long)l * 3 * kD,
            nullptr, 0, 0, 0,
            1.0f, kD, 1, 0, 0);

        // S = scale * Q @ K^T  (fp32, upper-diag tiles skipped)
        gemm_wmma_bf16<128, 128, false, false, 0, false>
            <<<dim3(kT / 128, kT / 128, kB * kH), 256, 0, stream>>>(
            qkvb,      3 * kD, (long)kT * 3 * kD, kHD,           // A = Q rows
            qkvb + kD, 3 * kD, (long)kT * 3 * kD, kHD,           // B = K rows ([key][hd] == [N][K])
            Sbuf, kT, (long)kH * kT * kT, (long)kT * kT,
            nullptr, nullptr, 0, 0, 0,
            0.125f, kHD, kH, /*causalSkip=*/1, 0);

        // P = causal softmax(S) -> bf16
        softmax_causal_kernel<<<kB * kH * kT, 256, 0, stream>>>(Sbuf, Pb);

        // Vt[b,h,hd,t] = V^T  (bf16)
        transpose_v_kernel<<<dim3(kT / 32, kHD / 32, kB * kH), 256, 0, stream>>>(qkvb, Vt);

        // O = P @ V  (K clipped at diagonal) -> bf16 [B,T,D]
        gemm_wmma_bf16<128, 64, false, false, 0, true>
            <<<dim3(kHD / 64, kT / 128, kB * kH), 256, 0, stream>>>(
            Pb, kT, (long)kH * kT * kT, (long)kT * kT,           // A = P
            Vt, kT, (long)kH * kHD * kT, (long)kHD * kT,         // B = V^T ([hd][t] == [N][K])
            obf, kD, (long)kT * kD, kHD,
            nullptr, nullptr, 0, 0, 0,
            1.0f, kT, kH, 0, /*clipK=*/1);

        // h = h + O @ proj_w + proj_b   (fp32)
        gemm_wmma_bf16<128, 128, true, true, 0, false>
            <<<dim3(kD / 128, BT / 128, 1), 256, 0, stream>>>(
            obf, kD, 0, 0,
            proj_wt + (long)l * kD * kD, kD, 0, 0,
            hbuf, kD, 0, 0,
            proj_b + (long)l * kD,
            hbuf, kD, 0, 0,
            1.0f, kD, 1, 0, 0);

        // u = LN2(h)
        layernorm_kernel<<<BT, 256, 0, stream>>>(
            hbuf, ln2_w + (long)l * kD, ln2_b + (long)l * kD, ubf);

        // ff = gelu(u @ ff1_w + ff1_b) -> bf16 [BT,DF]
        gemm_wmma_bf16<128, 128, true, false, 1, true>
            <<<dim3(kDF / 128, BT / 128, 1), 256, 0, stream>>>(
            ubf, kD, 0, 0,
            ff1_wt + (long)l * kDF * kD, kD, 0, 0,
            ffb, kDF, 0, 0,
            ff1_b + (long)l * kDF,
            nullptr, 0, 0, 0,
            1.0f, kD, 1, 0, 0);

        // h = h + ff @ ff2_w + ff2_b   (fp32)
        gemm_wmma_bf16<128, 128, true, true, 0, false>
            <<<dim3(kD / 128, BT / 128, 1), 256, 0, stream>>>(
            ffb, kDF, 0, 0,
            ff2_wt + (long)l * kD * kDF, kDF, 0, 0,
            hbuf, kD, 0, 0,
            ff2_b + (long)l * kD,
            hbuf, kD, 0, 0,
            1.0f, kDF, 1, 0, 0);
    }

    // logits = h @ W_out  -> fp32 [BT,V]
    cvt_f32_bf16_kernel<<<(kB * kT * kD) / 256, 256, 0, stream>>>(hbuf, hb16);
    gemm_wmma_bf16<128, 128, false, false, 0, false>
        <<<dim3(kV / 128, BT / 128, 1), 256, 0, stream>>>(
        hb16, kD, 0, 0,
        wout_t, kD, 0, 0,
        out, kV, 0, 0,
        nullptr, nullptr, 0, 0, 0,
        1.0f, kD, 1, 0, 0);
}
